// Attention_79963701117209
// MI455X (gfx1250) — matrix operand order repaired and checked by reading, not compiled, not run on hardware
//
#include <hip/hip_runtime.h>
#include <hip/hip_bf16.h>
#include <math.h>

typedef __attribute__((ext_vector_type(16))) __bf16 v16bf;
typedef __attribute__((ext_vector_type(8)))  __bf16 v8bf;
typedef __attribute__((ext_vector_type(2)))  __bf16 v2bf;
typedef __attribute__((ext_vector_type(8)))  float  v8f;
typedef int v4i __attribute__((vector_size(16)));   // matches builtin's param type

#define SEQ 2048
#define DM  4096
#define NH  32
#define HD  128

#if __has_builtin(__builtin_amdgcn_cvt_pk_bf16_f32)
#define HAS_PK_BF16 1
#else
#define HAS_PK_BF16 0
#endif

#if __has_builtin(__builtin_amdgcn_global_load_async_to_lds_b128)
#define HAS_ASYNC_LDS 1
#else
#define HAS_ASYNC_LDS 0
#endif

__device__ inline unsigned short f2bf(float f) {
    union { float f; unsigned u; } v; v.f = f;
    unsigned u = v.u;
    unsigned r = u + 0x7FFFu + ((u >> 16) & 1u);   // round-to-nearest-even
    return (unsigned short)(r >> 16);
}

__device__ inline unsigned pkbf(float a, float b) {
#if HAS_PK_BF16
    v2bf t = __builtin_amdgcn_cvt_pk_bf16_f32(a, b);
    union { v2bf v; unsigned u; } x; x.v = t; return x.u;
#else
    return (unsigned)f2bf(a) | ((unsigned)f2bf(b) << 16);
#endif
}

#if HAS_ASYNC_LDS
__device__ inline void async_cp16(const unsigned short* g, unsigned short* l) {
    __builtin_amdgcn_global_load_async_to_lds_b128(
        (__attribute__((address_space(1))) v4i*)g,
        (__attribute__((address_space(3))) v4i*)l, 0, 0);
}
__device__ inline void async_wait() {
    asm volatile("s_wait_asynccnt 0x0" ::: "memory");
}
#endif

// ---------------------------------------------------------------------------
// f32 -> bf16, 8 elements per thread, packed stores
// ---------------------------------------------------------------------------
__global__ __launch_bounds__(256)
void to_bf16(const float* __restrict__ src, unsigned short* __restrict__ dst) {
    size_t i = (size_t)(blockIdx.x * 256 + threadIdx.x) * 8;
    float4 a = *(const float4*)&src[i];
    float4 b = *(const float4*)&src[i + 4];
    uint4 o;
    o.x = pkbf(a.x, a.y); o.y = pkbf(a.z, a.w);
    o.z = pkbf(b.x, b.y); o.w = pkbf(b.z, b.w);
    *(uint4*)&dst[i] = o;
}

// ---------------------------------------------------------------------------
// C[m,n] = sum_k A[m,k] * W[n,k]   (A,W bf16 row-major, C f32)
// 128x128 tile / block, K-tile 64 (2 WMMA substeps per barrier pair),
// 8 waves, each wave 32(M)x64(N) = 8 wmma accumulators.
// ---------------------------------------------------------------------------
__global__ __launch_bounds__(256)
void gemm_bf16_xt(const unsigned short* __restrict__ A,
                  const unsigned short* __restrict__ W,
                  float* __restrict__ C, int M, int N, int K) {
    __shared__ unsigned short As[128 * 72];  // 64 k + pad (144B rows, 16B aligned)
    __shared__ unsigned short Ws[128 * 72];
    const int tid  = threadIdx.x;
    const int lane = tid & 31, wv = tid >> 5;
    const int l16  = lane & 15, g = lane >> 4;
    const int wm = wv & 3, wn = wv >> 2;           // 4x2 wave grid
    const int m0 = blockIdx.y * 128, n0 = blockIdx.x * 128;

    v8f acc[2][4];
    for (int a = 0; a < 2; a++) for (int b = 0; b < 4; b++)
        for (int e = 0; e < 8; e++) acc[a][b][e] = 0.f;

    for (int k0 = 0; k0 < K; k0 += 64) {
        // fill both 128x64 bf16 tiles (1024 16B chunks each)
#if HAS_ASYNC_LDS
        for (int i = 0; i < 4; i++) {
            int c = tid + i * 256;
            int row = c >> 3, col = (c & 7) << 3;
            async_cp16(&A[(size_t)(m0 + row) * K + k0 + col], &As[row * 72 + col]);
            async_cp16(&W[(size_t)(n0 + row) * K + k0 + col], &Ws[row * 72 + col]);
        }
        async_wait();
#else
        for (int i = 0; i < 4; i++) {
            int c = tid + i * 256;
            int row = c >> 3, col = (c & 7) << 3;
            *(uint4*)&As[row * 72 + col] = *(const uint4*)&A[(size_t)(m0 + row) * K + k0 + col];
            *(uint4*)&Ws[row * 72 + col] = *(const uint4*)&W[(size_t)(n0 + row) * K + k0 + col];
        }
#endif
        __syncthreads();

        for (int ks2 = 0; ks2 < 64; ks2 += 32) {
            v16bf afr[2], bfr[4];
            for (int mi = 0; mi < 2; mi++) {
                const unsigned short* rp = &As[(wm * 32 + mi * 16 + l16) * 72 + ks2];
                v8bf lo = *(const v8bf*)(rp + g * 8);
                v8bf hi = *(const v8bf*)(rp + 16 + g * 8);
                for (int e = 0; e < 8; e++) { afr[mi][e] = lo[e]; afr[mi][8 + e] = hi[e]; }
            }
            for (int ni = 0; ni < 4; ni++) {
                const unsigned short* rp = &Ws[(wn * 64 + ni * 16 + l16) * 72 + ks2];
                v8bf lo = *(const v8bf*)(rp + g * 8);
                v8bf hi = *(const v8bf*)(rp + 16 + g * 8);
                for (int e = 0; e < 8; e++) { bfr[ni][e] = lo[e]; bfr[ni][8 + e] = hi[e]; }
            }
            for (int mi = 0; mi < 2; mi++)
                for (int ni = 0; ni < 4; ni++)
                    acc[mi][ni] = __builtin_amdgcn_wmma_f32_16x16x32_bf16(
                        false, afr[mi], false, bfr[ni], (short)0, acc[mi][ni], false, false);
        }
        __syncthreads();
    }
    for (int mi = 0; mi < 2; mi++)
        for (int ni = 0; ni < 4; ni++)
            for (int j = 0; j < 8; j++) {
                int m = m0 + wm * 32 + mi * 16 + j + g * 8;
                int n = n0 + wn * 64 + ni * 16 + l16;
                C[(size_t)m * N + n] = acc[mi][ni][j];
            }
}

// ---------------------------------------------------------------------------
// RoPE on Q,K (fp32 math) + repack Q,K,V to bf16 [H][S][HD], packed stores.
// Each thread handles an adjacent pair (p, p+1) and its partner (p+64, p+65).
// ---------------------------------------------------------------------------
__global__ __launch_bounds__(256)
void rope_pack(const float* __restrict__ Qf, const float* __restrict__ Kf,
               const float* __restrict__ Vf, const float* __restrict__ cosb,
               const float* __restrict__ sinb, unsigned short* __restrict__ Qb,
               unsigned short* __restrict__ Kb, unsigned short* __restrict__ Vb) {
    int idx = blockIdx.x * 256 + threadIdx.x;  // S*H*32
    int p = (idx & 31) << 1;                   // 0,2,...,62
    int h = (idx >> 5) & (NH - 1);
    int s = idx >> 10;
    size_t base = (size_t)s * DM + h * HD;
    float2 c2 = *(const float2*)&cosb[s * HD + p];   // cos[p] == cos[p+64]
    float2 s2 = *(const float2*)&sinb[s * HD + p];
    float2 qa = *(const float2*)&Qf[base + p];
    float2 qb = *(const float2*)&Qf[base + p + 64];
    float2 ka = *(const float2*)&Kf[base + p];
    float2 kb = *(const float2*)&Kf[base + p + 64];
    float2 va = *(const float2*)&Vf[base + p];
    float2 vb = *(const float2*)&Vf[base + p + 64];
    size_t ob = ((size_t)h * SEQ + s) * HD;
    *(unsigned*)&Qb[ob + p]      = pkbf(qa.x * c2.x - qb.x * s2.x, qa.y * c2.y - qb.y * s2.y);
    *(unsigned*)&Qb[ob + p + 64] = pkbf(qb.x * c2.x + qa.x * s2.x, qb.y * c2.y + qa.y * s2.y);
    *(unsigned*)&Kb[ob + p]      = pkbf(ka.x * c2.x - kb.x * s2.x, ka.y * c2.y - kb.y * s2.y);
    *(unsigned*)&Kb[ob + p + 64] = pkbf(kb.x * c2.x + ka.x * s2.x, kb.y * c2.y + ka.y * s2.y);
    *(unsigned*)&Vb[ob + p]      = pkbf(va.x, va.y);
    *(unsigned*)&Vb[ob + p + 64] = pkbf(vb.x, vb.y);
}

// ---------------------------------------------------------------------------
// Flash attention: one (head, 128-row q block) per workgroup. 8 waves,
// each owns 16 query rows. Online softmax in f32, both GEMMs on bf16 WMMA.
// Dynamic LDS: Qs/Ks/Vt/Ps each 128x136 bf16 (stride 272B, bank-friendly).
// ---------------------------------------------------------------------------
__global__ __launch_bounds__(256)
void flash_attn(const unsigned short* __restrict__ Qb,
                const unsigned short* __restrict__ Kb,
                const unsigned short* __restrict__ Vb,
                float* __restrict__ attnO) {
    extern __shared__ unsigned short smem[];
    unsigned short* Qs = smem;            // 128*136
    unsigned short* Ks = smem + 17408;
    unsigned short* Vt = smem + 34816;    // transposed: [hd][key]
    unsigned short* Ps = smem + 52224;    // per-wave private rows
    const int tid = threadIdx.x, lane = tid & 31, wv = tid >> 5;
    const int l16 = lane & 15, g = lane >> 4;
    const int qb = blockIdx.x, h = blockIdx.y;
    const size_t headOff = (size_t)h * SEQ * HD;

    {   // load Q tile 128x128 bf16
        const unsigned short* src = Qb + headOff + (size_t)qb * 128 * HD;
        for (int i = 0; i < 8; i++) {
            int c = tid + i * 256;         // 2048 chunks of 8 bf16
            int row = c >> 4, c8 = c & 15;
#if HAS_ASYNC_LDS
            async_cp16(&src[row * HD + c8 * 8], &Qs[row * 136 + c8 * 8]);
#else
            *(uint4*)&Qs[row * 136 + c8 * 8] = *(const uint4*)&src[row * HD + c8 * 8];
#endif
        }
#if HAS_ASYNC_LDS
        async_wait();
#endif
    }

    v8f Oacc[8];
    for (int t = 0; t < 8; t++) for (int e = 0; e < 8; e++) Oacc[t][e] = 0.f;
    float curmax[8], lsum[8];
    for (int j = 0; j < 8; j++) { curmax[j] = -3.0e38f; lsum[j] = 0.f; }
    const float scale = 0.08838834764831845f;  // 1/sqrt(128)

    for (int kb = 0; kb <= qb; kb++) {
        __syncthreads();
        {   // load K tile (async) and V transposed (through VGPRs)
            const unsigned short* ksrc = Kb + headOff + (size_t)kb * 128 * HD;
            const unsigned short* vsrc = Vb + headOff + (size_t)kb * 128 * HD;
            for (int i = 0; i < 8; i++) {
                int c = tid + i * 256;
                int row = c >> 4, c8 = c & 15;
#if HAS_ASYNC_LDS
                async_cp16(&ksrc[row * HD + c8 * 8], &Ks[row * 136 + c8 * 8]);
#else
                *(uint4*)&Ks[row * 136 + c8 * 8] = *(const uint4*)&ksrc[row * HD + c8 * 8];
#endif
                uint4 v = *(const uint4*)&vsrc[row * HD + c8 * 8];
                const unsigned short* pv = (const unsigned short*)&v;
                for (int e = 0; e < 8; e++)
                    Vt[(c8 * 8 + e) * 136 + row] = pv[e];
            }
#if HAS_ASYNC_LDS
            async_wait();
#endif
        }
        __syncthreads();

        // S = Q K^T (16 rows x 128 keys per wave)
        v8f Sacc[8];
        for (int t = 0; t < 8; t++) for (int e = 0; e < 8; e++) Sacc[t][e] = 0.f;
        for (int ks = 0; ks < 4; ks++) {
            v16bf afr;
            {
                const unsigned short* rp = &Qs[(wv * 16 + l16) * 136 + ks * 32];
                v8bf lo = *(const v8bf*)(rp + g * 8);
                v8bf hi = *(const v8bf*)(rp + 16 + g * 8);
                for (int e = 0; e < 8; e++) { afr[e] = lo[e]; afr[8 + e] = hi[e]; }
            }
            for (int t = 0; t < 8; t++) {
                const unsigned short* rp = &Ks[(t * 16 + l16) * 136 + ks * 32];
                v8bf lo = *(const v8bf*)(rp + g * 8);
                v8bf hi = *(const v8bf*)(rp + 16 + g * 8);
                v16bf bfr;
                for (int e = 0; e < 8; e++) { bfr[e] = lo[e]; bfr[8 + e] = hi[e]; }
                Sacc[t] = __builtin_amdgcn_wmma_f32_16x16x32_bf16(
                    false, afr, false, bfr, (short)0, Sacc[t], false, false);
            }
        }
        // scale + causal mask (diagonal block only)
        const bool diag = (kb == qb);
        for (int t = 0; t < 8; t++)
            for (int j = 0; j < 8; j++) {
                float v = Sacc[t][j] * scale;
                if (diag) {
                    int m = wv * 16 + j + g * 8;
                    int n = t * 16 + l16;
                    if (n > m) v += -1000000000.0f;
                }
                Sacc[t][j] = v;
            }
        // online softmax; write P to per-wave LDS region
        for (int j = 0; j < 8; j++) {
            float tmax = Sacc[0][j];
            for (int t = 1; t < 8; t++) tmax = fmaxf(tmax, Sacc[t][j]);
            for (int o = 1; o < 16; o <<= 1) tmax = fmaxf(tmax, __shfl_xor(tmax, o, 32));
            float nm = fmaxf(curmax[j], tmax);
            float alpha = __expf(curmax[j] - nm);
            curmax[j] = nm;
            int mrow = wv * 16 + j + g * 8;
            float tsum = 0.f;
            for (int t = 0; t < 8; t++) {
                float p = __expf(Sacc[t][j] - nm);
                tsum += p;
                Ps[mrow * 136 + t * 16 + l16] = f2bf(p);
            }
            for (int o = 1; o < 16; o <<= 1) tsum += __shfl_xor(tsum, o, 32);
            lsum[j] = lsum[j] * alpha + tsum;
            for (int t = 0; t < 8; t++) Oacc[t][j] *= alpha;
        }
        // O += P * V (keys contracted, Vt contiguous per fragment)
        for (int ks = 0; ks < 4; ks++) {
            v16bf afr;
            {
                const unsigned short* rp = &Ps[(wv * 16 + l16) * 136 + ks * 32];
                v8bf lo = *(const v8bf*)(rp + g * 8);
                v8bf hi = *(const v8bf*)(rp + 16 + g * 8);
                for (int e = 0; e < 8; e++) { afr[e] = lo[e]; afr[8 + e] = hi[e]; }
            }
            for (int t = 0; t < 8; t++) {
                const unsigned short* rp = &Vt[(t * 16 + l16) * 136 + ks * 32];
                v8bf lo = *(const v8bf*)(rp + g * 8);
                v8bf hi = *(const v8bf*)(rp + 16 + g * 8);
                v16bf bfr;
                for (int e = 0; e < 8; e++) { bfr[e] = lo[e]; bfr[8 + e] = hi[e]; }
                Oacc[t] = __builtin_amdgcn_wmma_f32_16x16x32_bf16(
                    false, afr, false, bfr, (short)0, Oacc[t], false, false);
            }
        }
    }
    // normalize + write f32 [s][d]
    for (int t = 0; t < 8; t++)
        for (int j = 0; j < 8; j++) {
            int srow = qb * 128 + wv * 16 + j + g * 8;
            int hd = t * 16 + l16;
            attnO[(size_t)srow * DM + h * HD + hd] = Oacc[t][j] / lsum[j];
        }
}

// ---------------------------------------------------------------------------
extern "C" void kernel_launch(void* const* d_in, const int* in_sizes, int n_in,
                              void* d_out, int out_size, void* d_ws, size_t ws_size,
                              hipStream_t stream) {
    (void)in_sizes; (void)n_in; (void)out_size; (void)ws_size;
    const float* X    = (const float*)d_in[0];
    // d_in[1] = attention_mask (causal; applied analytically in flash_attn)
    const float* cosb = (const float*)d_in[2];
    const float* sinb = (const float*)d_in[3];
    const float* Wq   = (const float*)d_in[4];
    const float* Wk   = (const float*)d_in[5];
    const float* Wv   = (const float*)d_in[6];
    const float* Wo   = (const float*)d_in[7];
    float* out = (float*)d_out;

    char* ws = (char*)d_ws;
    const size_t SD = (size_t)SEQ * DM;       // 8M elements
    float* Qf = (float*)(ws);                 // 32MB
    float* Kf = (float*)(ws + SD * 4);
    float* Vf = (float*)(ws + SD * 8);
    float* Ao = (float*)(ws + SD * 12);
    unsigned short* Qbh = (unsigned short*)(ws + SD * 16);  // 16MB each
    unsigned short* Kbh = (unsigned short*)(ws + SD * 18);
    unsigned short* Vbh = (unsigned short*)(ws + SD * 20);
    unsigned short* Xb  = (unsigned short*)(ws + SD * 22);  // 16MB (reused for Ao bf16)
    unsigned short* Wqb = (unsigned short*)(ws + SD * 24);  // 32MB each
    unsigned short* Wkb = (unsigned short*)(ws + SD * 28);
    unsigned short* Wvb = (unsigned short*)(ws + SD * 32);
    unsigned short* Wob = (unsigned short*)(ws + SD * 36);  // total 40*SD = 320MB

    const int gX = (int)(SD / 2048);           // 4096 blocks (8 elems/thread)
    const int gW = (int)((size_t)DM * DM / 2048);
    to_bf16<<<gX, 256, 0, stream>>>(X,  Xb);
    to_bf16<<<gW, 256, 0, stream>>>(Wq, Wqb);
    to_bf16<<<gW, 256, 0, stream>>>(Wk, Wkb);
    to_bf16<<<gW, 256, 0, stream>>>(Wv, Wvb);
    to_bf16<<<gW, 256, 0, stream>>>(Wo, Wob);

    dim3 ggrid(DM / 128, SEQ / 128);
    gemm_bf16_xt<<<ggrid, 256, 0, stream>>>(Xb, Wqb, Qf, SEQ, DM, DM);
    gemm_bf16_xt<<<ggrid, 256, 0, stream>>>(Xb, Wkb, Kf, SEQ, DM, DM);
    gemm_bf16_xt<<<ggrid, 256, 0, stream>>>(Xb, Wvb, Vf, SEQ, DM, DM);

    rope_pack<<<(SEQ * NH * 32) / 256, 256, 0, stream>>>(Qf, Kf, Vf, cosb, sinb, Qbh, Kbh, Vbh);

    flash_attn<<<dim3(SEQ / 128, NH), 256, 4 * 128 * 136 * 2, stream>>>(Qbh, Kbh, Vbh, Ao);

    to_bf16<<<gX, 256, 0, stream>>>(Ao, Xb);   // reuse Xb for attn output bf16
    gemm_bf16_xt<<<ggrid, 256, 0, stream>>>(Xb, Wob, out, SEQ, DM, DM);
}
